// ContextualConv2d_59390807769180
// MI455X (gfx1250) — compile-verified, matrix operand order checked by reading, not run on hardware
//
#include <hip/hip_runtime.h>

// CDNA5 / gfx1250 implicit-GEMM 3x3 conv using v_wmma_f32_16x16x32_bf16 with
// bf16 hi/lo (split-2, 3-term) accumulation for ~fp32 accuracy.
//
// GEMM view: out[o, p] = sum_k W[o,k] * patch[k,p],  k = (kh,kw,cin), p = (n,h,w)
//  - A (weights) pre-swizzled into exact WMMA fragment order in workspace.
//  - B (im2col)  staged in LDS as bf16 hi/lo in [row][w][c] layout so each
//    fragment half is one ds_load_b128 (K runs of 8 contiguous channels).
//  - ctx[n,o] = c@c_weight^T + bias computed exactly in fp32, fused at store.

typedef __attribute__((ext_vector_type(16))) __bf16 v16bf;
typedef __attribute__((ext_vector_type(8)))  __bf16 v8bf;
typedef __attribute__((ext_vector_type(8)))  float  v8f;

namespace {
constexpr int Nn = 32, Ci = 128, Hh = 64, Ww = 64;
constexpr int Co = 256, Cd = 64;
constexpr int PIX  = 32;            // pixels (w) per workgroup
constexpr int LW   = PIX + 2;       // 34 LDS w-columns (left/right halo)
constexpr int NKS  = 36;            // 9 taps * (128/32) cin chunks
constexpr int NMT  = Co / 16;       // 16 output-channel tiles
constexpr int FRAG = 512;           // bf16 elements per 16x32 A fragment

constexpr size_t WHI_OFF   = 0;
constexpr size_t WHI_BYTES = (size_t)NKS * NMT * FRAG * 2;   // 589824 B
constexpr size_t WLO_OFF   = WHI_OFF + WHI_BYTES;
constexpr size_t CTX_OFF   = WLO_OFF + WHI_BYTES;            // + 32 KB ctx
} // namespace

// ---------------------------------------------------------------------------
// Pre-swizzle weights (OIHW fp32) into per-fragment bf16 hi/lo layout.
// Fragment id f = (tap*4 + kc)*16 + m_tile ; element order = lane*16 + e,
// matching the ISA 16-bit A layout:
//   lane<16 : M=lane,   K = {0..7 (v0..3), 16..23 (v4..7)}
//   lane>=16: M=lane-16,K = {8..15,        24..31}
// ---------------------------------------------------------------------------
__global__ __launch_bounds__(256) void prep_weights(
    const float* __restrict__ w, __bf16* __restrict__ whi,
    __bf16* __restrict__ wlo) {
  int idx  = blockIdx.x * 256 + threadIdx.x;       // 294912 total
  int frag = idx >> 9;
  int r    = idx & 511;
  int lane = r >> 4;
  int e    = r & 15;

  int mt  = frag % NMT;
  int tk  = frag / NMT;
  int kc  = tk % 4;
  int tap = tk / 4;
  int kh  = tap / 3, kw = tap % 3;

  int kk = ((e < 8) ? 0 : 16) + ((lane >= 16) ? 8 : 0) + (e & 7);
  int ci = kc * 32 + kk;
  int o  = mt * 16 + (lane & 15);

  float v  = w[((o * Ci + ci) * 3 + kh) * 3 + kw];
  __bf16 hi = (__bf16)v;
  __bf16 lo = (__bf16)(v - (float)hi);
  whi[idx] = hi;
  wlo[idx] = lo;
}

// ctx[n][o] = bias[o] + sum_d c[n][d] * c_weight[o][d]   (exact fp32)
__global__ __launch_bounds__(256) void prep_ctx(
    const float* __restrict__ c, const float* __restrict__ cw,
    const float* __restrict__ bias, float* __restrict__ ctx) {
  int n = blockIdx.x;       // 32
  int o = threadIdx.x;      // 256
  float s = bias[o];
  #pragma unroll 8
  for (int d = 0; d < Cd; ++d) s += c[n * Cd + d] * cw[o * Cd + d];
  ctx[n * Co + o] = s;
}

// ---------------------------------------------------------------------------
// Main conv kernel: one WG = all 256 out channels x 32 pixels of row (n,h).
// ---------------------------------------------------------------------------
__global__ __launch_bounds__(256) void conv_wmma(
    const float* __restrict__ x, const __bf16* __restrict__ whi,
    const __bf16* __restrict__ wlo, const float* __restrict__ ctx,
    float* __restrict__ out) {
  __shared__ __align__(16) __bf16 s_hi[3 * LW * Ci];   // 13056 bf16
  __shared__ __align__(16) __bf16 s_lo[3 * LW * Ci];
  __shared__ float s_ctx[Co];

  const int tid = threadIdx.x;
  const int b   = blockIdx.x;          // N*H*(W/PIX) = 4096
  const int pb  = b & 1;               // pixel half
  const int h   = (b >> 1) & (Hh - 1);
  const int n   = b >> 7;
  const int w0  = pb * PIX;

  // ---- LDS fill, interior columns (wl = 1..32): power-of-two decomposition,
  // w fastest -> coalesced global_load_b32, no integer div/mod.
  #pragma unroll 1
  for (int it = 0; it < 48; ++it) {                 // 3*128*32 / 256
    int idx = it * 256 + tid;
    int wofs = idx & 31;                            // w offset in tile
    int cc   = (idx >> 5) & (Ci - 1);
    int row  = idx >> 12;                           // 32*128 = 4096
    int hs   = h + row - 1;
    float v  = 0.0f;
    if (hs >= 0 && hs < Hh)
      v = x[((n * Ci + cc) * Hh + hs) * Ww + (w0 + wofs)];
    __bf16 hi = (__bf16)v;
    __bf16 lo = (__bf16)(v - (float)hi);
    int a = (row * LW + (wofs + 1)) * Ci + cc;
    s_hi[a] = hi;
    s_lo[a] = lo;
  }
  // ---- LDS fill, halo columns wl = 0 and wl = 33 (768 elements).
  #pragma unroll 1
  for (int it = 0; it < 3; ++it) {
    int idx  = it * 256 + tid;
    int cc   = idx & (Ci - 1);
    int side = (idx >> 7) & 1;
    int row  = idx >> 8;
    int wl   = side ? (LW - 1) : 0;
    int hs   = h + row - 1;
    int ws   = w0 + wl - 1;                         // w0-1 or w0+32
    float v  = 0.0f;
    if (hs >= 0 && hs < Hh && ws >= 0 && ws < Ww)
      v = x[((n * Ci + cc) * Hh + hs) * Ww + ws];
    __bf16 hi = (__bf16)v;
    __bf16 lo = (__bf16)(v - (float)hi);
    int a = (row * LW + wl) * Ci + cc;
    s_hi[a] = hi;
    s_lo[a] = lo;
  }
  if (tid < Co) s_ctx[tid] = ctx[n * Co + tid];
  __syncthreads();

  const int wave  = tid >> 5;
  const int lane  = tid & 31;
  const int col   = lane & 15;            // B column / D column (pixel)
  const int khalf = (lane >> 4) << 3;     // K-run offset 0 or 8

  v8f acc[2][2] = {};                     // [m-tile: wave, wave+8][pixel tile]

  #pragma unroll 1
  for (int tap = 0; tap < 9; ++tap) {
    const int kh = tap / 3, kw = tap % 3;
    #pragma unroll                        // 4x: lets scheduler overlap loads
    for (int kc = 0; kc < 4; ++kc) {      // with WMMA issue of prior chunk
      const int fbase = (tap * 4 + kc) * NMT;
      const int aoffA = (fbase + wave) * FRAG + lane * 16;
      const int aoffB = (fbase + wave + 8) * FRAG + lane * 16;
      const v16bf aA_hi = *(const v16bf*)(whi + aoffA);
      const v16bf aA_lo = *(const v16bf*)(wlo + aoffA);
      const v16bf aB_hi = *(const v16bf*)(whi + aoffB);
      const v16bf aB_lo = *(const v16bf*)(wlo + aoffB);
      const int cbase = kc * 32 + khalf;
      #pragma unroll
      for (int pt = 0; pt < 2; ++pt) {
        const int wlds = pt * 16 + col + kw;             // 0..33
        const int a0   = (kh * LW + wlds) * Ci + cbase;  // 16B aligned
        v8bf bh0 = *(const v8bf*)(s_hi + a0);
        v8bf bh1 = *(const v8bf*)(s_hi + a0 + 16);
        v8bf bl0 = *(const v8bf*)(s_lo + a0);
        v8bf bl1 = *(const v8bf*)(s_lo + a0 + 16);
        v16bf b_hi = __builtin_shufflevector(bh0, bh1, 0, 1, 2, 3, 4, 5, 6, 7,
                                             8, 9, 10, 11, 12, 13, 14, 15);
        v16bf b_lo = __builtin_shufflevector(bl0, bl1, 0, 1, 2, 3, 4, 5, 6, 7,
                                             8, 9, 10, 11, 12, 13, 14, 15);
        // D = A*B + C; split-2: hi*hi + hi*lo + lo*hi (~fp32 accuracy)
        acc[0][pt] = __builtin_amdgcn_wmma_f32_16x16x32_bf16(
            false, aA_hi, false, b_hi, (short)0, acc[0][pt], false, false);
        acc[0][pt] = __builtin_amdgcn_wmma_f32_16x16x32_bf16(
            false, aA_hi, false, b_lo, (short)0, acc[0][pt], false, false);
        acc[0][pt] = __builtin_amdgcn_wmma_f32_16x16x32_bf16(
            false, aA_lo, false, b_hi, (short)0, acc[0][pt], false, false);
        acc[1][pt] = __builtin_amdgcn_wmma_f32_16x16x32_bf16(
            false, aB_hi, false, b_hi, (short)0, acc[1][pt], false, false);
        acc[1][pt] = __builtin_amdgcn_wmma_f32_16x16x32_bf16(
            false, aB_hi, false, b_lo, (short)0, acc[1][pt], false, false);
        acc[1][pt] = __builtin_amdgcn_wmma_f32_16x16x32_bf16(
            false, aB_lo, false, b_hi, (short)0, acc[1][pt], false, false);
      }
    }
  }

  // Epilogue: D layout — VGPR v holds M = v (lanes<16) / v+8 (lanes>=16),
  // N = lane&15.  Fuse ctx (includes bias) and store fp32.
  const int osel = (lane < 16) ? 0 : 8;
  #pragma unroll
  for (int mi = 0; mi < 2; ++mi) {
    const int obase = (wave + mi * 8) * 16 + osel;
    #pragma unroll
    for (int pt = 0; pt < 2; ++pt) {
      const int w = w0 + pt * 16 + col;
      #pragma unroll
      for (int v = 0; v < 8; ++v) {
        const int o = obase + v;
        out[((n * Co + o) * Hh + h) * Ww + w] = acc[mi][pt][v] + s_ctx[o];
      }
    }
  }
}

// ---------------------------------------------------------------------------
extern "C" void kernel_launch(void* const* d_in, const int* in_sizes, int n_in,
                              void* d_out, int out_size, void* d_ws,
                              size_t ws_size, hipStream_t stream) {
  (void)in_sizes; (void)n_in; (void)out_size; (void)ws_size;
  const float* x    = (const float*)d_in[0];
  const float* c    = (const float*)d_in[1];
  const float* w    = (const float*)d_in[2];
  const float* cw   = (const float*)d_in[3];
  const float* bias = (const float*)d_in[4];
  float* out = (float*)d_out;

  char* ws = (char*)d_ws;   // needs ~1.16 MB: whi | wlo | ctx
  __bf16* whi = (__bf16*)(ws + WHI_OFF);
  __bf16* wlo = (__bf16*)(ws + WLO_OFF);
  float*  ctx = (float*)(ws + CTX_OFF);

  prep_weights<<<(NKS * NMT * FRAG) / 256, 256, 0, stream>>>(w, whi, wlo);
  prep_ctx<<<Nn, Co, 0, stream>>>(c, cw, bias, ctx);
  conv_wmma<<<Nn * Hh * (Ww / PIX), 256, 0, stream>>>(x, whi, wlo, ctx, out);
}
